// Timeline_83734682403453
// MI455X (gfx1250) — matrix operand order repaired and checked by reading, not compiled, not run on hardware
//
#include <hip/hip_runtime.h>

#define DEV __device__ __forceinline__

typedef __attribute__((ext_vector_type(16))) __bf16 v16bf;
typedef __attribute__((ext_vector_type(8)))  float  v8f;

union Frag16 { v16bf v; __bf16 e[16]; uint4 q[2]; };
union FragC  { v8f  v; float  e[8];  float4 f4[2]; };
union H8     { __bf16 h[8]; uint4 u; };

constexpr int Bn = 64, Vn = 64, Cn = 64, En = 256, Hn = 256, An = 128;
constexpr int VOCAB = 20000;
constexpr int SLD = 68;    // f32 pitch, scores [64 x 64]
constexpr int CLD = 260;   // f32 pitch, cell state [64 x 256]

DEV int laneid() { return threadIdx.x & 31; }
DEV float sigf(float x) { return 1.0f / (1.0f + __expf(-x)); }

// ---------------------------------------------------------------------------
// Fragment-native storage. One 16x32 (A) or 32x16 (B) bf16 tile = 512 elems.
// Element (lane, idx): first 8 idx at tile[lane*8 + idx], last 8 at
// tile[256 + lane*8 + (idx-8)]  -> each lane's fragment = two 16B chunks,
// 16B stride across lanes (conflict-free b128 loads).
// A-tile element (m,k): lane = (m&15) + 16*((k>>3)&1),
//                       idx  = (k&7) + 8*((k>>4)&1)        (k = k mod 32)
// B-tile element (k,n): lane = (n&15) + 16*((k>>3)&1), idx = (k&7) + 8*((k>>4)&1)
// C-fragment tile (16x16 f32) = 256 f32: lane's 8 values at tile*256+lane*8.
// ---------------------------------------------------------------------------
DEV int frag_off(int lane, int idx) {           // element offset within tile
  return ((idx & 8) << 5) + lane * 8 + (idx & 7);
}

DEV Frag16 load_frag(const __bf16* tile) {      // works for LDS and global
  Frag16 f;
  const char* p = (const char*)tile + laneid() * 16;
  f.q[0] = *(const uint4*)p;
  f.q[1] = *(const uint4*)(p + 512);
  return f;
}

// C/D 16x16 f32: VGPR r -> M = r + 8*(lane>=16), N = lane&15.
DEV void store_c_f32(float* dst, int ld, const FragC& c) {
  const int l = laneid();
  const int n  = l & 15;
  const int mb = (l >> 4) << 3;
#pragma unroll
  for (int r = 0; r < 8; ++r) dst[(mb + r)*ld + n] = c.e[r];
}

// Store a C tile (rows m0.., cols n0..) as A-fragments (frag grid has ktiles
// column-tiles of 32). Used for q -> scores-A.
DEV void store_c_as_afrag(__bf16* frag, int ktiles, int m0, int n0, const FragC& c) {
  const int l = laneid();
  const int k   = n0 + (l & 15);
  const int kt  = k >> 5;
  const int kl  = k & 31;
  const int idx = (kl & 7) + ((kl >> 4) << 3);
  const int lhi = ((kl >> 3) & 1) << 4;
#pragma unroll
  for (int r = 0; r < 8; ++r) {
    const int m = m0 + ((l >> 4) << 3) + r;
    __bf16* tile = frag + (size_t)((m >> 4) * ktiles + kt) * 512;
    tile[frag_off((m & 15) + lhi, idx)] = (__bf16)c.e[r];
  }
}

// Store a C tile as B-fragments: C element (m,n) becomes B element (k=n, n'=m).
// Used for k -> scores-B. frag grid: tile = kt*ntiles + nt.
DEV void store_c_as_bfrag(__bf16* frag, int ntiles, int m0, int n0, const FragC& c) {
  const int l = laneid();
  const int ks  = n0 + (l & 15);
  const int kt  = ks >> 5;
  const int idx = (ks & 7) + (((ks >> 4) & 1) << 3);
  const int lhi = ((ks >> 3) & 1) << 4;
#pragma unroll
  for (int r = 0; r < 8; ++r) {
    const int m = m0 + ((l >> 4) << 3) + r;
    __bf16* tile = frag + (size_t)(kt * ntiles + (m >> 4)) * 512;
    tile[frag_off((m & 15) + lhi, idx)] = (__bf16)c.e[r];
  }
}

DEV v8f wmma_bf16(const Frag16& a, const Frag16& b, v8f c) {
  return __builtin_amdgcn_wmma_f32_16x16x32_bf16(
      false, a.v, false, b.v, (short)0, c, false, false);
}

// Async global->LDS copy of one 16B chunk per lane (ASYNCcnt-tracked).
// LDS address = low 32 bits of the flat pointer.
DEV void async_copy16(void* lds_dst, const void* gsrc) {
  const unsigned loff = (unsigned)(uintptr_t)lds_dst;
  asm volatile("global_load_async_to_lds_b128 %0, %1, off"
               :: "v"(loff), "v"(gsrc) : "memory");
}
DEV void async_wait0() {
  asm volatile("s_wait_asynccnt 0x0" ::: "memory");
}

// ---------------------------------------------------------------------------
// Kernel 0: f32 -> bf16 conversion + B-fragment swizzle.
// dst element i: tile = i/512, off = i%512; inverse of frag_off gives
// (lane, idx) -> (k, n); src is [K][N] (cm=0) or [N][K] (cm=1).
// ---------------------------------------------------------------------------
__global__ void swizzleB_kernel(const float* __restrict__ src,
                                __bf16* __restrict__ dst,
                                int K, int N, int cm) {
  const int i = blockIdx.x * 256 + threadIdx.x;
  if (i >= K * N) return;
  const int tile = i >> 9;
  const int off  = i & 511;
  const int idx  = ((off >> 8) << 3) + (off & 7);
  const int lane = (off >> 3) & 31;
  const int NT = N >> 4;
  const int kt = tile / NT;
  const int nt = tile - kt * NT;
  const int k = (kt << 5) + ((lane >> 4) << 3) + (idx & 7) + ((idx >> 3) << 4);
  const int n = (nt << 4) + (lane & 15);
  const float v = cm ? src[(size_t)n * K + k] : src[(size_t)k * N + n];
  dst[i] = (__bf16)v;
}

// ---------------------------------------------------------------------------
// Kernel 1: fused per-visit attention + time-gated pooling.
// One block per (b,v) visit, 256 threads = 8 waves.
// vv[(v*B+b)][e] = ((gate*keep)^T softmax(mask(qk^T/sqrt(A)))) @ x
// vv is emitted in global A-fragment order for the Wih GEMM.
// ---------------------------------------------------------------------------
__global__ void attn_kernel(const int* __restrict__ codes,
                            const float* __restrict__ mask,
                            const float* __restrict__ times,
                            const float* __restrict__ emb,
                            const __bf16* __restrict__ wqf,
                            const __bf16* __restrict__ wkf,
                            const float* __restrict__ decay,
                            const float* __restrict__ initial,
                            __bf16* __restrict__ vvfrag) {
  extern __shared__ char smem[];
  __bf16* xfrag = (__bf16*)smem;                 // 4mt x 8kt x 512 = 32768 B
  __bf16* qfrag = (__bf16*)(smem + 32768);       // 4mt x 4kt x 512 = 16384 B
  __bf16* kfrag = (__bf16*)(smem + 49152);       // 4kt x 4nt x 512 = 16384 B
  float*  sc    = (float*) (smem + 65536);       // [64][SLD]        17408 B
  float*  gk    = (float*) (smem + 82944);       // [64]
  float*  wv    = (float*) (smem + 83200);       // [64]
  int*    lcode = (int*)   (smem + 83456);       // [64]

  const int n   = blockIdx.x;     // n = b*V + v
  const int b   = n / Vn;
  const int v   = n - b * Vn;
  const int tid = threadIdx.x;
  const float t = times[n];

  if (tid < Cn) {
    const int code = codes[(size_t)n * Cn + tid];
    lcode[tid] = code;
    const float keep = (code == 0) ? 0.f : 1.f;
    gk[tid] = sigf(decay[code] * t + initial[code]) * keep;
  }
  __syncthreads();

  // Gather embedding rows -> bf16 A-fragments; 8 consecutive k = one 16B store
  {
    const int c  = tid >> 2;
    const int e0 = (tid & 3) * 64;
    const float* src = emb + (size_t)lcode[c] * En + e0;
    const int mt = c >> 4;
    const int ml = c & 15;
#pragma unroll
    for (int g = 0; g < 8; ++g) {
      const int k = e0 + g * 8;                  // 8-aligned
      const float4 f0 = *(const float4*)(src + g * 8);
      const float4 f1 = *(const float4*)(src + g * 8 + 4);
      H8 h8;
      h8.h[0] = (__bf16)f0.x; h8.h[1] = (__bf16)f0.y;
      h8.h[2] = (__bf16)f0.z; h8.h[3] = (__bf16)f0.w;
      h8.h[4] = (__bf16)f1.x; h8.h[5] = (__bf16)f1.y;
      h8.h[6] = (__bf16)f1.z; h8.h[7] = (__bf16)f1.w;
      const int lane_ = ml + (((k >> 3) & 1) << 4);
      __bf16* tile = xfrag + (size_t)(mt * 8 + (k >> 5)) * 512;
      *(uint4*)(tile + ((k & 16) ? 256 : 0) + lane_ * 8) = h8.u;
    }
  }
  __syncthreads();

  const int wave = tid >> 5;

  // q = x@WQ, k = x@WK : 64 tiles of 16x16, K=256.
  for (int tt = wave; tt < 64; tt += 8) {
    const bool isK = tt >= 32;
    const int t2 = tt & 31;
    const int m0 = (t2 >> 3) * 16;
    const int n0 = (t2 & 7) * 16;
    const __bf16* wb = isK ? wkf : wqf;
    FragC acc;
#pragma unroll
    for (int r = 0; r < 8; ++r) acc.e[r] = 0.f;
    for (int kt = 0; kt < 8; ++kt) {
      Frag16 a  = load_frag(xfrag + (size_t)((m0 >> 4) * 8 + kt) * 512);
      Frag16 bb = load_frag(wb + (size_t)(kt * 8 + (n0 >> 4)) * 512);
      acc.v = wmma_bf16(a, bb, acc.v);
    }
    if (isK) store_c_as_bfrag(kfrag, 4, m0, n0, acc);   // scores B (k^T)
    else     store_c_as_afrag(qfrag, 4, m0, n0, acc);   // scores A
  }
  __syncthreads();

  // scores = q @ k^T : 16 tiles, K=128.
  for (int tt = wave; tt < 16; tt += 8) {
    const int m0 = (tt >> 2) * 16;
    const int n0 = (tt & 3) * 16;
    FragC acc;
#pragma unroll
    for (int r = 0; r < 8; ++r) acc.e[r] = 0.f;
    for (int kt = 0; kt < 4; ++kt) {
      Frag16 a  = load_frag(qfrag + (size_t)((m0 >> 4) * 4 + kt) * 512);
      Frag16 bb = load_frag(kfrag + (size_t)(kt * 4 + (n0 >> 4)) * 512);
      acc.v = wmma_bf16(a, bb, acc.v);
    }
    store_c_f32(sc + m0 * SLD + n0, SLD, acc);
  }
  __syncthreads();

  // Masked softmax per row (row-mask term is softmax-invariant, omitted).
  if (tid < Cn) {
    const float scale = 0.08838834764831845f;  // 1/sqrt(128)
    const float* mrow = mask + (size_t)n * Cn;
    float* row = sc + tid * SLD;
    float mx = -3.4e38f;
    for (int c = 0; c < Cn; ++c) {
      const float val = (row[c] - mrow[c]) * scale;
      row[c] = val;
      mx = fmaxf(mx, val);
    }
    float s = 0.f;
    for (int c = 0; c < Cn; ++c) { float e = __expf(row[c] - mx); row[c] = e; s += e; }
    const float inv = 1.f / s;
    for (int c = 0; c < Cn; ++c) row[c] *= inv;
  }
  __syncthreads();

  // w = (gate*keep)^T @ s
  if (tid < Cn) {
    float s = 0.f;
    for (int c = 0; c < Cn; ++c) s += gk[c] * sc[c * SLD + tid];
    wv[tid] = s;
  }
  __syncthreads();

  // vv[e] = w @ x, written to global vvfrag in A-fragment order.
  {
    const int e = tid;
    const int kt = e >> 5;
    const int idx = (e & 7) + (((e >> 4) & 1) << 3);
    const int lhi = ((e >> 3) & 1) << 4;
    float s = 0.f;
    for (int c = 0; c < Cn; ++c) {
      const __bf16* tile = xfrag + (size_t)((c >> 4) * 8 + kt) * 512;
      s += wv[c] * (float)tile[frag_off((c & 15) + lhi, idx)];
    }
    const int m = v * Bn + b;                    // global row in [4096 x 256]
    __bf16* tile = vvfrag + (size_t)((m >> 4) * 8 + kt) * 512;
    tile[frag_off((m & 15) + lhi, idx)] = (__bf16)s;
  }
}

// ---------------------------------------------------------------------------
// Kernel 2: hoisted input-projection GEMM for both LSTM directions.
// xz[dir] = VV @ Wih^T + b  :  M=4096, N=1024, K=256.
// A panel (contiguous 32KB of vvfrag) is async-copied to LDS (8x reuse);
// B from swizzled Wih fragments in L2. Output in C-fragment order.
// ---------------------------------------------------------------------------
__global__ void xz_gemm_kernel(const __bf16* __restrict__ vvfrag,
                               const __bf16* __restrict__ wihf,
                               const __bf16* __restrict__ wihb,
                               const float* __restrict__ biasf,
                               const float* __restrict__ biasb,
                               float* __restrict__ xzf,
                               float* __restrict__ xzb) {
  extern __shared__ char smem[];
  __bf16* apanel = (__bf16*)smem;          // 32 tiles x 512 = 32768 B

  const __bf16* wih = blockIdx.z ? wihb : wihf;
  const float* bias = blockIdx.z ? biasb : biasf;
  float* xz = blockIdx.z ? xzb : xzf;
  const int n0g = blockIdx.y * 128;
  const int tid = threadIdx.x;

  // Async-stage the 64x256 A panel: rows [bx*64, bx*64+64) of vvfrag are the
  // contiguous element range [bx*16384, bx*16384+16384).
  {
    const char* gsrc = (const char*)(vvfrag + (size_t)blockIdx.x * 16384);
#pragma unroll
    for (int i = 0; i < 8; ++i) {
      const int c = tid + i * 256;         // 2048 x 16B = 32768 B
      async_copy16((char*)apanel + c * 16, gsrc + c * 16);
    }
    async_wait0();
  }
  __syncthreads();

  const int wave = tid >> 5;
  const int l = tid & 31;
  const int ln = l & 15;

  for (int i = 0; i < 4; ++i) {
    const int tt = wave * 4 + i;             // 32 tiles: 4 m x 8 n
    const int m0 = (tt >> 3) * 16;
    const int n0 = (tt & 7) * 16;
    const int ntg = (n0g + n0) >> 4;
    FragC acc;
#pragma unroll
    for (int r = 0; r < 8; ++r) acc.e[r] = 0.f;
    for (int kt = 0; kt < 8; ++kt) {
      Frag16 a  = load_frag(apanel + (size_t)((m0 >> 4) * 8 + kt) * 512);
      Frag16 bb = load_frag(wih + (size_t)(kt * 64 + ntg) * 512);
      acc.v = wmma_bf16(a, bb, acc.v);
    }
    const float bv = bias[n0g + n0 + ln];
#pragma unroll
    for (int r = 0; r < 8; ++r) acc.e[r] += bv;
    // store in C-fragment order: tile = mt_glob*64 + nt_glob, lane's 8 floats
    const int mtg = blockIdx.x * 4 + (m0 >> 4);
    float* dst = xz + ((size_t)(mtg * 64 + ntg) << 8) + l * 8;
    *(float4*)dst       = acc.f4[0];
    *(float4*)(dst + 4) = acc.f4[1];
  }
}

// ---------------------------------------------------------------------------
// Kernel 3: sequential LSTM recurrence; one persistent workgroup per
// direction (grid=2), 512 threads = 16 waves. c-state f32 + h-state
// A-fragments in LDS; gate-aligned tiles so each wave fuses i/f/g/o.
// xz is read in C-fragment order (two float4 per gate-tile per lane).
// Both directions record h at visit V-1 (reference uses hs_b[-1]).
// ---------------------------------------------------------------------------
__global__ void lstm_rec_kernel(const float* __restrict__ xzf,
                                const float* __restrict__ xzb,
                                const __bf16* __restrict__ whhf,
                                const __bf16* __restrict__ whhb,
                                float* __restrict__ hfin) {
  extern __shared__ char smem[];
  float*  cst   = (float*)smem;                     // [64][CLD] f32  66560 B
  __bf16* hfrag = (__bf16*)(smem + Bn * CLD * 4);   // 4mt x 8kt x 512, 32768 B

  const int dir = blockIdx.x;
  const float* xz = dir ? xzb : xzf;
  const __bf16* whh = dir ? whhb : whhf;
  const int tid = threadIdx.x;
  const int wave = tid >> 5;
  const int l = tid & 31;
  const int ln = l & 15, lm = (l >> 4) << 3;

  for (int i = tid; i < Bn * CLD; i += blockDim.x) cst[i] = 0.f;
  for (int i = tid; i < 4 * 8 * 512; i += blockDim.x) hfrag[i] = (__bf16)0.f;
  __syncthreads();

  for (int t = 0; t < Vn; ++t) {
    const int tv = dir ? (Vn - 1 - t) : t;

    FragC acc[16];
#pragma unroll
    for (int q = 0; q < 16; ++q)
#pragma unroll
      for (int r = 0; r < 8; ++r) acc[q].e[r] = 0.f;

    // GEMM phase: z_tiles = h @ Whh^T (reads previous h fragments)
    for (int i = 0; i < 4; ++i) {
      const int g = wave * 4 + i;         // 64 groups = 4 m-tiles x 16 j-tiles
      const int m0 = (g >> 4) * 16;
      const int j0 = (g & 15) * 16;
      for (int kt = 0; kt < 8; ++kt) {
        Frag16 a = load_frag(hfrag + (size_t)((m0 >> 4) * 8 + kt) * 512);
#pragma unroll
        for (int u = 0; u < 4; ++u) {     // gates i,f,g,o at N = u*256 + j0
          Frag16 bb = load_frag(whh + (size_t)(kt * 64 + u * 16 + (j0 >> 4)) * 512);
          acc[i * 4 + u].v = wmma_bf16(a, bb, acc[i * 4 + u].v);
        }
      }
    }
    __syncthreads();

    // Gate + state update phase: each (m,j) patch owned by exactly one wave.
    for (int i = 0; i < 4; ++i) {
      const int g = wave * 4 + i;
      const int m0 = (g >> 4) * 16;
      const int j0 = (g & 15) * 16;
      const int j = j0 + ln;
      const int jkt = j >> 5;
      const int jl = j & 31;
      const int jidx = (jl & 7) + ((jl >> 4) << 3);
      const int jlhi = ((jl >> 3) & 1) << 4;
      // xz tiles in C-fragment order; same (lane,r) mapping as our acc
      FragC xg[4];
      const int mtg = tv * 4 + (m0 >> 4);
#pragma unroll
      for (int u = 0; u < 4; ++u) {
        const float* base = xz + ((size_t)(mtg * 64 + u * 16 + (j0 >> 4)) << 8) + l * 8;
        xg[u].f4[0] = *(const float4*)base;
        xg[u].f4[1] = *(const float4*)(base + 4);
      }
#pragma unroll
      for (int r = 0; r < 8; ++r) {
        const int m = m0 + lm + r;
        const float zi = acc[i * 4 + 0].e[r] + xg[0].e[r];
        const float zf = acc[i * 4 + 1].e[r] + xg[1].e[r];
        const float zg = acc[i * 4 + 2].e[r] + xg[2].e[r];
        const float zo = acc[i * 4 + 3].e[r] + xg[3].e[r];
        const float cOld = cst[m * CLD + j];
        const float cNew = sigf(zf) * cOld + sigf(zi) * tanhf(zg);
        const float h = sigf(zo) * tanhf(cNew);
        cst[m * CLD + j] = cNew;
        __bf16* tile = hfrag + (size_t)((m >> 4) * 8 + jkt) * 512;
        tile[frag_off((m & 15) + jlhi, jidx)] = (__bf16)h;
        if (tv == Vn - 1)
          hfin[(size_t)dir * Bn * Hn + (size_t)m * Hn + j] = h;
      }
    }
    __syncthreads();
  }
}

// ---------------------------------------------------------------------------
// Kernel 4: output GEMM + sigmoid. out = sigmoid([h_f|h_b] @ W_out + b_out)
// M=64, K=512, N=20000. 1250 n-tiles, 8 per block -> 157 blocks.
// ---------------------------------------------------------------------------
__global__ void out_gemm_kernel(const float* __restrict__ hfin,
                                const __bf16* __restrict__ woutf,
                                const float* __restrict__ bout,
                                float* __restrict__ out) {
  extern __shared__ char smem[];
  __bf16* afrag = (__bf16*)smem;                    // 4mt x 16kt x 512, 65536 B
  const int tid = threadIdx.x;

  // Stage concat(h_f, h_b) as A-fragments (8 consecutive k -> one 16B store)
  {
    const int r  = tid >> 2;
    const int k0 = (tid & 3) * 128;
    const int mt = r >> 4;
    const int ml = r & 15;
#pragma unroll
    for (int g = 0; g < 16; ++g) {
      const int k = k0 + g * 8;
      const float* src = (k < Hn)
          ? hfin + (size_t)r * Hn + k
          : hfin + (size_t)Bn * Hn + (size_t)r * Hn + (k - Hn);
      H8 h8;
#pragma unroll
      for (int j = 0; j < 8; ++j) h8.h[j] = (__bf16)src[j];
      const int lane_ = ml + (((k >> 3) & 1) << 4);
      __bf16* tile = afrag + (size_t)(mt * 16 + (k >> 5)) * 512;
      *(uint4*)(tile + ((k & 16) ? 256 : 0) + lane_ * 8) = h8.u;
    }
  }
  __syncthreads();

  const int wave = tid >> 5;
  const int l = tid & 31;
  const int ln = l & 15, lm = (l >> 4) << 3;
  const int ntile = blockIdx.x * 8 + wave;
  if (ntile >= VOCAB / 16) return;
  const int n0 = ntile * 16;

  for (int mt = 0; mt < 4; ++mt) {
    FragC acc;
#pragma unroll
    for (int r = 0; r < 8; ++r) acc.e[r] = 0.f;
    for (int kt = 0; kt < 16; ++kt) {
      Frag16 a  = load_frag(afrag + (size_t)(mt * 16 + kt) * 512);
      Frag16 bb = load_frag(woutf + (size_t)(kt * 1250 + ntile) * 512);
      acc.v = wmma_bf16(a, bb, acc.v);
    }
    const float bv = bout[n0 + ln];
#pragma unroll
    for (int r = 0; r < 8; ++r) {
      const int m = mt * 16 + lm + r;
      out[(size_t)m * VOCAB + n0 + ln] = sigf(acc.e[r] + bv);
    }
  }
}

// ---------------------------------------------------------------------------
extern "C" void kernel_launch(void* const* d_in, const int* in_sizes, int n_in,
                              void* d_out, int out_size, void* d_ws, size_t ws_size,
                              hipStream_t stream) {
  const int*   codes   = (const int*)  d_in[0];
  const float* mask    = (const float*)d_in[1];
  const float* times   = (const float*)d_in[2];
  const float* emb     = (const float*)d_in[3];
  const float* WQ      = (const float*)d_in[4];
  const float* WK      = (const float*)d_in[5];
  const float* decay   = (const float*)d_in[6];
  const float* initial = (const float*)d_in[7];
  const float* WihF    = (const float*)d_in[8];
  const float* WhhF    = (const float*)d_in[9];
  const float* bF      = (const float*)d_in[10];
  const float* WihB    = (const float*)d_in[11];
  const float* WhhB    = (const float*)d_in[12];
  const float* bB      = (const float*)d_in[13];
  const float* Wout    = (const float*)d_in[14];
  const float* bout    = (const float*)d_in[15];
  float* out = (float*)d_out;

  char* ws = (char*)d_ws;
  size_t off = 0;
  auto carve = [&](size_t bytes) -> char* {
    char* p = ws + off;
    off += (bytes + 255) & ~(size_t)255;
    return p;
  };
  __bf16* wqf    = (__bf16*)carve((size_t)En * An * 2);
  __bf16* wkf    = (__bf16*)carve((size_t)En * An * 2);
  __bf16* wihff  = (__bf16*)carve((size_t)4 * Hn * En * 2);
  __bf16* whhff  = (__bf16*)carve((size_t)4 * Hn * Hn * 2);
  __bf16* wihbf  = (__bf16*)carve((size_t)4 * Hn * En * 2);
  __bf16* whhbf  = (__bf16*)carve((size_t)4 * Hn * Hn * 2);
  __bf16* woutf  = (__bf16*)carve((size_t)2 * Hn * VOCAB * 2);
  __bf16* vvfrag = (__bf16*)carve((size_t)Vn * Bn * En * 2);
  float*  xzf    = (float*) carve((size_t)Vn * Bn * 4 * Hn * 4);
  float*  xzb    = (float*) carve((size_t)Vn * Bn * 4 * Hn * 4);
  float*  hfin   = (float*) carve((size_t)2 * Bn * Hn * 4);

  auto swz = [&](const float* s, __bf16* d, int K, int N, int cm) {
    swizzleB_kernel<<<(K * N + 255) / 256, 256, 0, stream>>>(s, d, K, N, cm);
  };
  swz(WQ,   wqf,   En, An, 0);          // [E][A] row-major = [K][N]
  swz(WK,   wkf,   En, An, 0);
  swz(WihF, wihff, En, 4 * Hn, 1);      // B(k,n) = Wih[n][k]
  swz(WhhF, whhff, Hn, 4 * Hn, 1);
  swz(WihB, wihbf, En, 4 * Hn, 1);
  swz(WhhB, whhbf, Hn, 4 * Hn, 1);
  swz(Wout, woutf, 2 * Hn, VOCAB, 0);   // [2H][VOCAB] = [K][N]

  // Stage 1: fused attention + pooling (4096 visits)
  attn_kernel<<<Bn * Vn, 256, 83712, stream>>>(
      codes, mask, times, emb, wqf, wkf, decay, initial, vvfrag);

  // Stage 2a: hoisted Wih GEMMs for both directions (async A staging)
  xz_gemm_kernel<<<dim3(64, 8, 2), 256, 32768, stream>>>(
      vvfrag, wihff, wihbf, bF, bB, xzf, xzb);

  // Stage 2b: sequential recurrence, one persistent WG per direction
  lstm_rec_kernel<<<2, 512, (size_t)Bn * CLD * 4 + 4 * 8 * 512 * 2, stream>>>(
      xzf, xzb, whhff, whhbf, hfin);

  // Stage 3: output projection + sigmoid
  out_gemm_kernel<<<(VOCAB / 16 + 7) / 8, 256, 65536, stream>>>(
      hfin, woutf, bout, out);
}
